// PointNetSetAbstraction_60275571032494
// MI455X (gfx1250) — compile-verified, matrix operand order checked by reading, not run on hardware
//
#include <hip/hip_runtime.h>

typedef __attribute__((ext_vector_type(16))) _Float16 v16h;
typedef __attribute__((ext_vector_type(8)))  _Float16 v8h;
typedef __attribute__((ext_vector_type(8)))  float    v8f;

#define B_  16
#define N_  16384
#define S_  512
#define K_  32
#define M_  (B_ * S_ * K_)   /* 262144 grouped positions */
#define R2_ 0.04f            /* RADIUS^2 */

// ---------------------------------------------------------------------------
// Furthest point sampling: one 1024-thread block per batch. Each thread owns
// 16 points; running min-distance lives in registers, argmax via LDS tree.
// ---------------------------------------------------------------------------
__global__ __launch_bounds__(1024) void fps_kernel(const float* __restrict__ xyz,
                                                   int* __restrict__ cents)
{
    const int b = blockIdx.x;
    const int t = threadIdx.x;
    const float* px = xyz + (size_t)b * 3 * N_;
    const float* py = px + N_;
    const float* pz = py + N_;

    float x[16], y[16], z[16], d[16];
#pragma unroll
    for (int i = 0; i < 16; ++i) {
        const int n = t + i * 1024;
        x[i] = px[n]; y[i] = py[n]; z[i] = pz[n];
        d[i] = 1e10f;
    }
    __shared__ float rv[1024];
    __shared__ int   ri[1024];
    int far = 0;
    for (int it = 0; it < S_; ++it) {
        if (t == 0) cents[b * S_ + it] = far;
        const float cx = px[far], cy = py[far], cz = pz[far];
        float best = -1.0f; int bi = 0;
#pragma unroll
        for (int i = 0; i < 16; ++i) {
            const int n = t + i * 1024;
            const float dx = x[i] - cx, dy = y[i] - cy, dz = z[i] - cz;
            const float dd = dx * dx + dy * dy + dz * dz;
            d[i] = fminf(d[i], dd);
            if (d[i] > best) { best = d[i]; bi = n; }
        }
        rv[t] = best; ri[t] = bi;
        __syncthreads();
        for (int off = 512; off > 0; off >>= 1) {
            if (t < off) {
                const float ov = rv[t + off]; const int oi = ri[t + off];
                if (ov > rv[t] || (ov == rv[t] && oi < ri[t])) { rv[t] = ov; ri[t] = oi; }
            }
            __syncthreads();
        }
        far = ri[0];
        __syncthreads();
    }
}

// ---------------------------------------------------------------------------
// Ball query + grouping: one wave32 per centroid. Ballot + prefix-popcount
// reproduces "first K in-radius indices ascending, pad with first hit".
// Writes X0 directly as f16 rows [M, 32] (channels 0-8 real, 9-31 zero pad)
// and the new_xyz output [B,3,S].
// ---------------------------------------------------------------------------
__global__ __launch_bounds__(256) void group_kernel(const float* __restrict__ xyz,
                                                    const float* __restrict__ pts,
                                                    const int* __restrict__ cents,
                                                    _Float16* __restrict__ X0,
                                                    float* __restrict__ out_newxyz)
{
    const int gw   = (int)((blockIdx.x * 256u + threadIdx.x) >> 5);  // centroid id
    const int lane = threadIdx.x & 31;
    const int b = gw >> 9, s = gw & (S_ - 1);

    const float* px = xyz + (size_t)b * 3 * N_;
    const float* py = px + N_;
    const float* pz = py + N_;
    const float* pp = pts + (size_t)b * 6 * N_;

    const int ci = cents[b * S_ + s];
    const float cx = px[ci], cy = py[ci], cz = pz[ci];
    if (lane == 0) {
        out_newxyz[(size_t)b * 3 * S_ + 0 * S_ + s] = cx;
        out_newxyz[(size_t)b * 3 * S_ + 1 * S_ + s] = cy;
        out_newxyz[(size_t)b * 3 * S_ + 2 * S_ + s] = cz;
    }

    int cnt = 0, first = -1;
    const size_t mbase = (size_t)gw * K_;
    for (int base = 0; base < N_ && cnt < K_; base += 32) {
        const int n = base + lane;
        const float dx = px[n] - cx, dy = py[n] - cy, dz = pz[n] - cz;
        const float dd = dx * dx + dy * dy + dz * dz;
        const bool hit = dd <= R2_;
        const unsigned mask = (unsigned)__ballot(hit ? 1 : 0);
        if (first < 0 && mask != 0u) first = base + __ffs(mask) - 1;
        const int pos = cnt + __popc(mask & ((1u << lane) - 1u));
        if (hit && pos < K_) {
            _Float16* row = X0 + (mbase + pos) * 32;
            row[0] = (_Float16)dx; row[1] = (_Float16)dy; row[2] = (_Float16)dz;
#pragma unroll
            for (int c = 0; c < 6; ++c) row[3 + c] = (_Float16)pp[(size_t)c * N_ + n];
#pragma unroll
            for (int c = 9; c < 32; ++c) row[c] = (_Float16)0.f;
        }
        cnt += __popc(mask);
    }
    if (cnt > K_) cnt = K_;
    if (cnt < K_) {
        const int f = (first >= 0) ? first : (N_ - 1);
        const float dx = px[f] - cx, dy = py[f] - cy, dz = pz[f] - cz;
        float fp[6];
#pragma unroll
        for (int c = 0; c < 6; ++c) fp[c] = pp[(size_t)c * N_ + f];
        for (int pos = cnt + lane; pos < K_; pos += 32) {
            _Float16* row = X0 + (mbase + pos) * 32;
            row[0] = (_Float16)dx; row[1] = (_Float16)dy; row[2] = (_Float16)dz;
#pragma unroll
            for (int c = 0; c < 6; ++c) row[3 + c] = (_Float16)fp[c];
#pragma unroll
            for (int c = 9; c < 32; ++c) row[c] = (_Float16)0.f;
        }
    }
}

// ---------------------------------------------------------------------------
// Weight prep: f32 -> f16, pad W0 9->32 columns.
// ---------------------------------------------------------------------------
__global__ void prep_weights_kernel(const float* __restrict__ W0,
                                    const float* __restrict__ W1,
                                    const float* __restrict__ W2,
                                    _Float16* __restrict__ W0h,
                                    _Float16* __restrict__ W1h,
                                    _Float16* __restrict__ W2h)
{
    const int t = blockIdx.x * blockDim.x + threadIdx.x;
    if (t < 64 * 32) {
        const int r = t >> 5, c = t & 31;
        W0h[t] = (c < 9) ? (_Float16)W0[r * 9 + c] : (_Float16)0.f;
    }
    if (t < 64 * 64)  W1h[t] = (_Float16)W1[t];
    if (t < 128 * 64) W2h[t] = (_Float16)W2[t];
}

__global__ void zero_stats_kernel(float* __restrict__ p, int n)
{
    const int t = blockIdx.x * blockDim.x + threadIdx.x;
    if (t < n) p[t] = 0.f;
}

// ---------------------------------------------------------------------------
// WMMA GEMM:  Y[M,OC] = W[OC,ICP] * X[M,ICP]^T  (+bias), f16 in / f32 acc.
// Optionally applies per-channel BN(scale,shift)+ReLU to X on load.
// Accumulates per-channel sum / sum-of-squares (LDS ds_add_f32 -> one
// global_atomic_add_f32 per channel per block) for the next layer's BN.
// Each wave computes a 16-column strip for all OC tiles; B-fragments are
// loaded once per wave as two contiguous v8h (16B) chunks per K tile,
// matching the CDNA5 16-bit 16x32 fragment layout.
// ---------------------------------------------------------------------------
template<int ICP, int OC, bool BN>
__global__ __launch_bounds__(256) void gemm_bn_kernel(
    const _Float16* __restrict__ X, const _Float16* __restrict__ Wh,
    const float* __restrict__ bias,
    const float* __restrict__ scale, const float* __restrict__ shift,
    _Float16* __restrict__ Y, float* __restrict__ gsum, float* __restrict__ gsq)
{
    __shared__ float s_sum[OC];
    __shared__ float s_sq[OC];
    for (int i = threadIdx.x; i < OC; i += 256) { s_sum[i] = 0.f; s_sq[i] = 0.f; }
    __syncthreads();

    const int wave  = threadIdx.x >> 5;
    const int lane  = threadIdx.x & 31;
    const int half  = lane >> 4;        // 0: K 0-7/16-23, 1: K 8-15/24-31
    const int lcol  = lane & 15;
    const int col   = (blockIdx.x * 8 + wave) * 16 + lcol;
    const int khalf = half * 8;

    // B fragments (activations), reused across all OC tiles.
    v16h bfrag[ICP / 32];
    const _Float16* xrow = X + (size_t)col * ICP;
#pragma unroll
    for (int kt = 0; kt < ICP / 32; ++kt) {
        v8h lo = *(const v8h*)(xrow + kt * 32 + khalf);
        v8h hi = *(const v8h*)(xrow + kt * 32 + khalf + 16);
        if constexpr (BN) {
#pragma unroll
            for (int j = 0; j < 8; ++j) {
                const int c0 = kt * 32 + khalf + j;
                const int c1 = c0 + 16;
                const float xl = (float)lo[j] * scale[c0] + shift[c0];
                const float xh = (float)hi[j] * scale[c1] + shift[c1];
                lo[j] = (_Float16)fmaxf(xl, 0.f);
                hi[j] = (_Float16)fmaxf(xh, 0.f);
            }
        }
        v16h bb;
#pragma unroll
        for (int j = 0; j < 8; ++j) { bb[j] = lo[j]; bb[8 + j] = hi[j]; }
        bfrag[kt] = bb;
    }

#pragma unroll
    for (int ot = 0; ot < OC / 16; ++ot) {
        v8f acc = {};
#pragma unroll
        for (int kt = 0; kt < ICP / 32; ++kt) {
            const _Float16* wrow = Wh + (size_t)(ot * 16 + lcol) * ICP + kt * 32 + khalf;
            const v8h alo = *(const v8h*)wrow;
            const v8h ahi = *(const v8h*)(wrow + 16);
            v16h aa;
#pragma unroll
            for (int j = 0; j < 8; ++j) { aa[j] = alo[j]; aa[8 + j] = ahi[j]; }
            acc = __builtin_amdgcn_wmma_f32_16x16x32_f16(false, aa, false, bfrag[kt],
                                                         (short)0, acc, false, false);
        }
        const int rbase = ot * 16 + half * 8;   // C/D layout: VGPR r -> row r (+8 hi lanes)
#pragma unroll
        for (int r = 0; r < 8; ++r) {
            const int oc = rbase + r;
            const float yv = acc[r] + bias[oc];
            Y[(size_t)col * OC + oc] = (_Float16)yv;
            atomicAdd(&s_sum[oc], yv);
            atomicAdd(&s_sq[oc], yv * yv);
        }
    }
    __syncthreads();
    for (int i = threadIdx.x; i < OC; i += 256) {
        atomicAdd(&gsum[i], s_sum[i]);
        atomicAdd(&gsq[i], s_sq[i]);
    }
}

// Fold (mean, rstd, gamma, beta) into per-channel scale/shift.
__global__ void finalize_kernel(const float* __restrict__ sum, const float* __restrict__ sq,
                                const float* __restrict__ g, const float* __restrict__ be,
                                float* __restrict__ scale, float* __restrict__ shift, int C)
{
    const int c = threadIdx.x;
    if (c < C) {
        const float invM = 1.0f / (float)M_;
        const float m  = sum[c] * invM;
        const float v  = sq[c] * invM - m * m;
        const float rs = rsqrtf(v + 1e-5f);
        const float sc = rs * g[c];
        scale[c] = sc;
        shift[c] = be[c] - m * sc;
    }
}

// BN+ReLU on layer-3 output, max over K=32, write pooled [B,128,S].
__global__ __launch_bounds__(128) void pool_kernel(const _Float16* __restrict__ Y3,
                                                   const float* __restrict__ scale,
                                                   const float* __restrict__ shift,
                                                   float* __restrict__ out)
{
    const int bs = blockIdx.x;        // b*S + s
    const int oc = threadIdx.x;       // 0..127
    const int b = bs >> 9, s = bs & (S_ - 1);
    const _Float16* p = Y3 + (size_t)bs * K_ * 128 + oc;
    const float sc = scale[oc], sh = shift[oc];
    float m = 0.f;                    // ReLU output is >= 0
#pragma unroll
    for (int k = 0; k < K_; ++k) {
        float x = (float)p[(size_t)k * 128] * sc + sh;
        x = fmaxf(x, 0.f);
        m = fmaxf(m, x);
    }
    out[(size_t)b * 128 * S_ + (size_t)oc * S_ + s] = m;
}

// ---------------------------------------------------------------------------
extern "C" void kernel_launch(void* const* d_in, const int* in_sizes, int n_in,
                              void* d_out, int out_size, void* d_ws, size_t ws_size,
                              hipStream_t stream)
{
    (void)in_sizes; (void)n_in; (void)out_size; (void)ws_size;

    const float* xyz = (const float*)d_in[0];
    const float* pts = (const float*)d_in[1];
    const float* W0  = (const float*)d_in[2];
    const float* b0  = (const float*)d_in[3];
    const float* g0  = (const float*)d_in[4];
    const float* be0 = (const float*)d_in[5];
    const float* W1  = (const float*)d_in[6];
    const float* b1  = (const float*)d_in[7];
    const float* g1  = (const float*)d_in[8];
    const float* be1 = (const float*)d_in[9];
    const float* W2  = (const float*)d_in[10];
    const float* b2  = (const float*)d_in[11];
    const float* g2  = (const float*)d_in[12];
    const float* be2 = (const float*)d_in[13];
    float* out = (float*)d_out;

    // Workspace carve-up (256B aligned slices).
    char* w = (char*)d_ws;
    auto carve = [&](size_t bytes) -> void* {
        void* p = (void*)w;
        w += (bytes + 255) & ~(size_t)255;
        return p;
    };
    int*      cents  = (int*)     carve((size_t)B_ * S_ * sizeof(int));
    float*    stats  = (float*)   carve(512 * sizeof(float));          // s0,q0,s1,q1,s2,q2
    float*    scale0 = (float*)   carve(64 * sizeof(float));
    float*    shift0 = (float*)   carve(64 * sizeof(float));
    float*    scale1 = (float*)   carve(64 * sizeof(float));
    float*    shift1 = (float*)   carve(64 * sizeof(float));
    float*    scale2 = (float*)   carve(128 * sizeof(float));
    float*    shift2 = (float*)   carve(128 * sizeof(float));
    _Float16* W0h    = (_Float16*)carve((size_t)64 * 32 * 2);
    _Float16* W1h    = (_Float16*)carve((size_t)64 * 64 * 2);
    _Float16* W2h    = (_Float16*)carve((size_t)128 * 64 * 2);
    _Float16* X0     = (_Float16*)carve((size_t)M_ * 32 * 2);          // 16 MB
    _Float16* Y1     = (_Float16*)carve((size_t)M_ * 64 * 2);          // 32 MB
    _Float16* Y2     = (_Float16*)carve((size_t)M_ * 64 * 2);          // 32 MB
    _Float16* Y3     = (_Float16*)carve((size_t)M_ * 128 * 2);         // 64 MB

    zero_stats_kernel<<<2, 256, 0, stream>>>(stats, 512);
    prep_weights_kernel<<<32, 256, 0, stream>>>(W0, W1, W2, W0h, W1h, W2h);
    fps_kernel<<<B_, 1024, 0, stream>>>(xyz, cents);
    group_kernel<<<(B_ * S_) / 8, 256, 0, stream>>>(xyz, pts, cents, X0, out);

    gemm_bn_kernel<32, 64, false><<<M_ / 128, 256, 0, stream>>>(
        X0, W0h, b0, nullptr, nullptr, Y1, stats + 0, stats + 64);
    finalize_kernel<<<1, 64, 0, stream>>>(stats + 0, stats + 64, g0, be0, scale0, shift0, 64);

    gemm_bn_kernel<64, 64, true><<<M_ / 128, 256, 0, stream>>>(
        Y1, W1h, b1, scale0, shift0, Y2, stats + 128, stats + 192);
    finalize_kernel<<<1, 64, 0, stream>>>(stats + 128, stats + 192, g1, be1, scale1, shift1, 64);

    gemm_bn_kernel<64, 128, true><<<M_ / 128, 256, 0, stream>>>(
        Y2, W2h, b2, scale1, shift1, Y3, stats + 256, stats + 384);
    finalize_kernel<<<1, 128, 0, stream>>>(stats + 256, stats + 384, g2, be2, scale2, shift2, 128);

    pool_kernel<<<B_ * S_, 128, 0, stream>>>(Y3, scale2, shift2, out + (size_t)B_ * 3 * S_);
}
